// MyDeeperGCN_36490042147518
// MI455X (gfx1250) — matrix operand order repaired and checked by reading, not compiled, not run on hardware
//
#include <hip/hip_runtime.h>
#include <hip/hip_bf16.h>

typedef __attribute__((ext_vector_type(16))) _Float16 v16h;
typedef __attribute__((ext_vector_type(8)))  float    v8f;

#define BATCH 16
#define SEQ   107
#define NPAD  107
#define NROWS 214        // SEQ + NPAD
#define NNODE 3424       // BATCH * NROWS
#define NEDGE 12000
#define HDIM  64

// ---- WMMA fragment K-index maps (contiguous k = e + 16*kh, same for A and B; 16-bit operands, wave32) ----
__device__ __forceinline__ int kmapA(int e, int kh) {   // A: 16x32, lane row = l&15
    return e + 16 * kh;
}
__device__ __forceinline__ int kmapB(int e, int kh) {   // B: 32x16, lane col = l&15
    return e + 16 * kh;
}

// =========================================================================
// Pre-permute w2/b2 into WMMA B-fragment order (runs once per launch).
// W2B : [4 layers][64 i-slabs][4 n-tiles][32 lanes][16 elems] f16
// B2rB: [4 layers][2 k-slabs][4 n-tiles][32 lanes][16 elems] f16
// =========================================================================
__global__ void build_w2b_kernel(const float* __restrict__ w2,   // [4][4096][32]
                                 const float* __restrict__ b2,   // [4][4096]
                                 _Float16* __restrict__ W2B,
                                 _Float16* __restrict__ B2rB) {
    int idx = blockIdx.x * blockDim.x + threadIdx.x;
    const int totalW = 4 * 64 * 4 * 32 * 16;   // 524288
    const int totalB = 4 * 2 * 4 * 32 * 16;    // 16384
    if (idx < totalW) {
        int e    = idx & 15;
        int lane = (idx >> 4) & 31;
        int nt   = (idx >> 9) & 3;
        int i    = (idx >> 11) & 63;
        int layer= idx >> 17;
        int kh = lane >> 4, m = lane & 15;
        int k   = kmapB(e, kh);
        int col = nt * 16 + m;
        // B_slab_i[k][col] = w2[i*64+col, k]
        W2B[idx] = (_Float16)w2[((size_t)layer * 4096 + i * 64 + col) * 32 + k];
    } else if (idx < totalW + totalB) {
        int j = idx - totalW;
        int e    = j & 15;
        int lane = (j >> 4) & 31;
        int nt   = (j >> 9) & 3;
        int s    = (j >> 11) & 1;
        int layer= j >> 12;
        int kh = lane >> 4, m = lane & 15;
        int row = 32 * s + kmapB(e, kh);
        int col = nt * 16 + m;
        // B2r[i][o] = b2[i*64+o]
        B2rB[j] = (_Float16)b2[layer * 4096 + row * 64 + col];
    }
}

// =========================================================================
// Fused NNConv message kernel: one wave per 16 edges.
// msg(16x64) = z(16x2048) @ W2p + x(16x64) @ B2r ; scatter-add into agg.
// z slab i is formed in-register: a[e] = x[m,i] * ew[m, kA(e)].
// B fragments are double-buffered so each WMMA waits on loads issued one
// slab earlier (s_wait_loadcnt <= 8) instead of a full L2 round trip.
// =========================================================================
#define LOADB(d0, d1, d2, d3, i)                                   \
    do {                                                           \
        const _Float16* wp = wbase + (size_t)(i) * 2048;           \
        d0 = *(const v16h*)(wp);                                   \
        d1 = *(const v16h*)(wp + 512);                             \
        d2 = *(const v16h*)(wp + 1024);                            \
        d3 = *(const v16h*)(wp + 1536);                            \
    } while (0)

#define WMMA4(a, b0, b1, b2, b3)                                                                   \
    do {                                                                                           \
        acc0 = __builtin_amdgcn_wmma_f32_16x16x32_f16(false, a, false, b0, (short)0, acc0, false, false); \
        acc1 = __builtin_amdgcn_wmma_f32_16x16x32_f16(false, a, false, b1, (short)0, acc1, false, false); \
        acc2 = __builtin_amdgcn_wmma_f32_16x16x32_f16(false, a, false, b2, (short)0, acc2, false, false); \
        acc3 = __builtin_amdgcn_wmma_f32_16x16x32_f16(false, a, false, b3, (short)0, acc3, false, false); \
    } while (0)

__global__ __launch_bounds__(32) void nnconv_msg_kernel(
    const float* __restrict__ t,        // [N,64] source node features
    const float* __restrict__ ew,       // [E,32] per-edge embedding @ w1
    const int*   __restrict__ src,
    const int*   __restrict__ dst,
    const _Float16* __restrict__ W2B,   // this layer's slab fragments
    const _Float16* __restrict__ B2rB,
    float* __restrict__ agg)            // [N,64]
{
    __shared__ float xs[16 * 64];
    __shared__ float ews[16 * 32];
    __shared__ int   sdst[16];
    __shared__ int   ssrc[16];

    const int lane = threadIdx.x;
    const int e0   = blockIdx.x * 16;
    if (lane < 16) { ssrc[lane] = src[e0 + lane]; sdst[lane] = dst[e0 + lane]; }
    __syncthreads();
    for (int i = lane; i < 16 * 64; i += 32) {
        int r = i >> 6, c = i & 63;
        xs[i] = t[(size_t)ssrc[r] * 64 + c];
    }
    for (int i = lane; i < 16 * 32; i += 32) {
        int r = i >> 5, c = i & 31;
        ews[i] = ew[(size_t)(e0 + r) * 32 + c];
    }
    __syncthreads();

    const int m = lane & 15, kh = lane >> 4;
    const _Float16* wbase = W2B + (size_t)lane * 16;

    // per-lane ew fragment (A K-order), reused for all 64 slabs
    v16h ewf;
#pragma unroll
    for (int e = 0; e < 16; ++e) ewf[e] = (_Float16)ews[m * 32 + kmapA(e, kh)];

    v8f acc0 = {}, acc1 = {}, acc2 = {}, acc3 = {};
    v16h pa0, pa1, pa2, pa3;   // B fragments, buffer A (even slabs)
    v16h pb0, pb1, pb2, pb3;   // B fragments, buffer B (odd slabs)

    LOADB(pa0, pa1, pa2, pa3, 0);
    for (int i = 0; i < 64; i += 2) {
        // prefetch odd slab while computing even slab
        LOADB(pb0, pb1, pb2, pb3, i + 1);
        {
            _Float16 xsc = (_Float16)xs[m * 64 + i];
            v16h a;
#pragma unroll
            for (int e = 0; e < 16; ++e) a[e] = ewf[e] * xsc;
            WMMA4(a, pa0, pa1, pa2, pa3);
        }
        // prefetch next even slab while computing odd slab
        if (i + 2 < 64) LOADB(pa0, pa1, pa2, pa3, i + 2);
        {
            _Float16 xsc = (_Float16)xs[m * 64 + i + 1];
            v16h a;
#pragma unroll
            for (int e = 0; e < 16; ++e) a[e] = ewf[e] * xsc;
            WMMA4(a, pb0, pb1, pb2, pb3);
        }
    }

    // bias GEMM: x(16x64) @ B2r(64x64), 2 K-slabs
    {
        const _Float16* wb2 = B2rB + (size_t)lane * 16;
        LOADB(pa0, pa1, pa2, pa3, 0 + (wb2 - wbase) / 2048);   // keep types simple below instead
    }
#pragma unroll
    for (int s = 0; s < 2; ++s) {
        const _Float16* bb = B2rB + (size_t)(s * 4) * 512 + (size_t)lane * 16;
        v16h bf0 = *(const v16h*)(bb);
        v16h bf1 = *(const v16h*)(bb + 512);
        v16h bf2 = *(const v16h*)(bb + 1024);
        v16h bf3 = *(const v16h*)(bb + 1536);
        v16h a;
#pragma unroll
        for (int e = 0; e < 16; ++e) a[e] = (_Float16)xs[m * 64 + 32 * s + kmapA(e, kh)];
        WMMA4(a, bf0, bf1, bf2, bf3);
    }

    // D layout: VGPR r -> edge row r + 8*kh, col = m (+16*nt)
#pragma unroll
    for (int r = 0; r < 8; ++r) {
        int row = r + 8 * kh;
        float* ap = agg + (size_t)sdst[row] * 64;
        atomicAdd(ap +  0 + m, acc0[r]);
        atomicAdd(ap + 16 + m, acc1[r]);
        atomicAdd(ap + 32 + m, acc2[r]);
        atomicAdd(ap + 48 + m, acc3[r]);
    }
}

// =========================================================================
// BiLSTM: one block per (batch, direction); 128 threads = 128 gates.
// =========================================================================
__global__ __launch_bounds__(128) void lstm_kernel(
    const float* __restrict__ xin, int b_rows, int IW,
    const float* __restrict__ Wih, const float* __restrict__ Whh,
    const float* __restrict__ bih, const float* __restrict__ bhh,
    float* __restrict__ out)        // [B,SEQ,64], dir in cols [dir*32, dir*32+32)
{
    int b   = blockIdx.x >> 1;
    int dir = blockIdx.x & 1;
    int j   = threadIdx.x;
    __shared__ float h[32], c[32], g[128];
    if (j < 32) { h[j] = 0.f; c[j] = 0.f; }
    __syncthreads();
    const float* wih = Wih + (size_t)dir * 128 * IW;
    const float* whh = Whh + (size_t)dir * 128 * 32;
    float bias = bih[dir * 128 + j] + bhh[dir * 128 + j];
    for (int tt = 0; tt < SEQ; ++tt) {
        int tpos = dir ? (SEQ - 1 - tt) : tt;
        const float* xr = xin + ((size_t)b * b_rows + tpos) * IW;
        float acc = bias;
        for (int k = 0; k < IW; ++k) acc += xr[k] * wih[j * IW + k];
        for (int k = 0; k < 32; ++k) acc += h[k] * whh[j * 32 + k];
        g[j] = acc;
        __syncthreads();
        if (j < 32) {
            float ig = 1.f / (1.f + __expf(-g[j]));
            float fg = 1.f / (1.f + __expf(-g[32 + j]));
            float gg = tanhf(g[64 + j]);
            float og = 1.f / (1.f + __expf(-g[96 + j]));
            float cn = fg * c[j] + ig * gg;
            c[j] = cn;
            float hn = og * tanhf(cn);
            h[j] = hn;
            out[((size_t)b * SEQ + tpos) * 64 + dir * 32 + j] = hn;
        }
        __syncthreads();
    }
}

// initial node features -> t: nuc rows from lstm1, pad rows via node MLP
__global__ void assemble_kernel(const float* __restrict__ x,
                                const float* __restrict__ h_l,
                                const float* __restrict__ mw,
                                const float* __restrict__ mb,
                                float* __restrict__ t) {
    int id = blockIdx.x * blockDim.x + threadIdx.x;
    if (id >= NNODE * 64) return;
    int n = id >> 6, o = id & 63;
    int b = n / NROWS, r = n - b * NROWS;
    float v;
    if (r < SEQ) {
        v = h_l[((size_t)b * SEQ + r) * 64 + o];
    } else {
        v = mb[o];
        const float* xr = x + (size_t)n * 14;
        for (int f = 0; f < 14; ++f) v += xr[f] * mw[o * 14 + f];
    }
    t[id] = v;
}

// generic out[n,j] = relu?( b[j] + in[n,:] . W[j,:] )   (W stored [J,K])
__global__ void linear_kernel(const float* __restrict__ in, const float* __restrict__ W,
                              const float* __restrict__ bias, float* __restrict__ out,
                              int rows, int K, int J, int relu) {
    int id = blockIdx.x * blockDim.x + threadIdx.x;
    if (id >= rows * J) return;
    int n = id / J, j = id - n * J;
    float acc = bias[j];
    const float* ir = in + (size_t)n * K;
    const float* wr = W + (size_t)j * K;
    for (int k = 0; k < K; ++k) acc += ir[k] * wr[k];
    if (relu) acc = fmaxf(acc, 0.f);
    out[id] = acc;
}

// h[n,o] = residual*h[n,o] + agg[n,o] + t[n,:] . root[:,o] + bias[o]
__global__ void combine_kernel(const float* __restrict__ t, const float* __restrict__ agg,
                               const float* __restrict__ root, const float* __restrict__ bias,
                               float* __restrict__ h, int residual) {
    int id = blockIdx.x * blockDim.x + threadIdx.x;
    if (id >= NNODE * 64) return;
    int n = id >> 6, o = id & 63;
    float acc = agg[id] + bias[o];
    const float* tr = t + (size_t)n * 64;
    for (int i = 0; i < 64; ++i) acc += tr[i] * root[i * 64 + o];
    h[id] = residual ? (h[id] + acc) : acc;
}

// LayerNorm(+optional ReLU) per 64-wide row; one wave per row
__global__ __launch_bounds__(32) void ln_relu_kernel(const float* __restrict__ in,
                                                     const float* __restrict__ g,
                                                     const float* __restrict__ b,
                                                     float* __restrict__ out, int relu) {
    int row = blockIdx.x, l = threadIdx.x;
    float x0 = in[(size_t)row * 64 + l], x1 = in[(size_t)row * 64 + 32 + l];
    float s = x0 + x1;
    for (int o = 16; o > 0; o >>= 1) s += __shfl_xor(s, o, 32);
    float mean = s * (1.f / 64.f);
    float d0 = x0 - mean, d1 = x1 - mean;
    float v = d0 * d0 + d1 * d1;
    for (int o = 16; o > 0; o >>= 1) v += __shfl_xor(v, o, 32);
    float rs = rsqrtf(v * (1.f / 64.f) + 1e-5f);
    float y0 = d0 * rs * g[l] + b[l];
    float y1 = d1 * rs * g[32 + l] + b[32 + l];
    if (relu) { y0 = fmaxf(y0, 0.f); y1 = fmaxf(y1, 0.f); }
    out[(size_t)row * 64 + l] = y0;
    out[(size_t)row * 64 + 32 + l] = y1;
}

__global__ void gather_nuc_kernel(const float* __restrict__ t, float* __restrict__ ht) {
    int id = blockIdx.x * blockDim.x + threadIdx.x;
    if (id >= BATCH * SEQ * 64) return;
    int o = id & 63, n = id >> 6;
    int b = n / SEQ, s = n - b * SEQ;
    ht[id] = t[((size_t)b * NROWS + s) * 64 + o];
}

__global__ void add_kernel(const float* __restrict__ a, const float* __restrict__ b,
                           float* __restrict__ out, int n) {
    int id = blockIdx.x * blockDim.x + threadIdx.x;
    if (id < n) out[id] = a[id] + b[id];
}

// attention per reference's einsum: l,m range over BATCH (16), batched over (s,head)
__global__ void attn_kernel(const float* __restrict__ qkv, float* __restrict__ o) {
    int id = blockIdx.x * blockDim.x + threadIdx.x;
    if (id >= SEQ * 4 * 16) return;
    int l = id & 15;
    int hh = (id >> 4) & 3;
    int bs = id >> 6;
    float q[16];
#pragma unroll
    for (int d = 0; d < 16; ++d)
        q[d] = qkv[((size_t)l * SEQ + bs) * 192 + hh * 16 + d] * 0.25f;
    float sc[16];
    float mx = -1e30f;
#pragma unroll
    for (int mm = 0; mm < 16; ++mm) {
        float acc = 0.f;
        const float* kr = qkv + ((size_t)mm * SEQ + bs) * 192 + 64 + hh * 16;
#pragma unroll
        for (int d = 0; d < 16; ++d) acc += q[d] * kr[d];
        sc[mm] = acc;
        mx = fmaxf(mx, acc);
    }
    float sum = 0.f;
#pragma unroll
    for (int mm = 0; mm < 16; ++mm) { sc[mm] = __expf(sc[mm] - mx); sum += sc[mm]; }
    float inv = 1.f / sum;
#pragma unroll
    for (int d = 0; d < 16; ++d) {
        float acc = 0.f;
#pragma unroll
        for (int mm = 0; mm < 16; ++mm)
            acc += sc[mm] * qkv[((size_t)mm * SEQ + bs) * 192 + 128 + hh * 16 + d];
        o[((size_t)l * SEQ + bs) * 64 + hh * 16 + d] = acc * inv;
    }
}

// =========================================================================
extern "C" void kernel_launch(void* const* d_in, const int* in_sizes, int n_in,
                              void* d_out, int out_size, void* d_ws, size_t ws_size,
                              hipStream_t stream) {
    const float* x         = (const float*)d_in[0];
    const float* edge_attr = (const float*)d_in[1];
    const int*   edge_idx  = (const int*)  d_in[2];
    // d_in[3]=nuc_idx, d_in[4]=nonnuc_idx (index structure reproduced analytically)
    const float* l1_wih = (const float*)d_in[5];
    const float* l1_whh = (const float*)d_in[6];
    const float* l1_bih = (const float*)d_in[7];
    const float* l1_bhh = (const float*)d_in[8];
    const float* mlp_w  = (const float*)d_in[9];
    const float* mlp_b  = (const float*)d_in[10];
    const float* edge_w = (const float*)d_in[11];
    const float* edge_b = (const float*)d_in[12];
    const float* nn1_w  = (const float*)d_in[13];
    const float* nn1_b  = (const float*)d_in[14];
    const float* nn2_w  = (const float*)d_in[15];
    const float* nn2_b  = (const float*)d_in[16];
    const float* root_w = (const float*)d_in[17];
    const float* gcn_b  = (const float*)d_in[18];
    const float* ln_g   = (const float*)d_in[19];
    const float* ln_b   = (const float*)d_in[20];
    const float* inpj_w = (const float*)d_in[21];
    const float* inpj_b = (const float*)d_in[22];
    const float* outp_w = (const float*)d_in[23];
    const float* outp_b = (const float*)d_in[24];
    const float* ln1_g  = (const float*)d_in[25];
    const float* ln1_b  = (const float*)d_in[26];
    const float* ln2_g  = (const float*)d_in[27];
    const float* ln2_b  = (const float*)d_in[28];
    const float* ff1_w  = (const float*)d_in[29];
    const float* ff1_b  = (const float*)d_in[30];
    const float* ff2_w  = (const float*)d_in[31];
    const float* ff2_b  = (const float*)d_in[32];
    const float* l2_wih = (const float*)d_in[33];
    const float* l2_whh = (const float*)d_in[34];
    const float* l2_bih = (const float*)d_in[35];
    const float* l2_bhh = (const float*)d_in[36];
    const float* lin_w  = (const float*)d_in[37];
    const float* lin_b  = (const float*)d_in[38];
    float* outp = (float*)d_out;

    const int* src = edge_idx;
    const int* dst = edge_idx + NEDGE;

    // ---- workspace bump allocator ----
    char* p = (char*)d_ws;
    auto bump = [&](size_t bytes) -> char* {
        char* r = p;
        p += (bytes + 255) & ~(size_t)255;
        return r;
    };
    float* h_l  = (float*)bump((size_t)BATCH * SEQ * 64 * 4);
    float* tbuf = (float*)bump((size_t)NNODE * 64 * 4);
    float* hbuf = (float*)bump((size_t)NNODE * 64 * 4);
    float* agg  = (float*)bump((size_t)NNODE * 64 * 4);
    float* eemb = (float*)bump((size_t)NEDGE * 32 * 4);
    float* ew   = (float*)bump((size_t)NEDGE * 32 * 4);
    _Float16* W2B  = (_Float16*)bump((size_t)4 * 64 * 4 * 32 * 16 * 2);
    _Float16* B2rB = (_Float16*)bump((size_t)4 * 2 * 4 * 32 * 16 * 2);
    float* h_t  = (float*)bump((size_t)BATCH * SEQ * 64 * 4);
    float* qkv  = (float*)bump((size_t)BATCH * SEQ * 192 * 4);
    float* obuf = (float*)bump((size_t)BATCH * SEQ * 64 * 4);
    float* tmp1 = (float*)bump((size_t)BATCH * SEQ * 64 * 4);
    float* tmp2 = (float*)bump((size_t)BATCH * SEQ * 64 * 4);
    float* hs2  = (float*)bump((size_t)BATCH * SEQ * 64 * 4);
    float* h2   = (float*)bump((size_t)BATCH * SEQ * 64 * 4);
    (void)ws_size; (void)in_sizes; (void)n_in; (void)out_size;

    const int NS = BATCH * SEQ;           // 1712
    const int N64 = NNODE * 64;           // 219136
    auto cdiv = [](int a, int b) { return (a + b - 1) / b; };

    // 1. BiLSTM #1 over nucleotide rows (rows b*214+s, width 14)
    lstm_kernel<<<BATCH * 2, 128, 0, stream>>>(x, NROWS, 14, l1_wih, l1_whh, l1_bih, l1_bhh, h_l);
    // 2. assemble initial node features into tbuf
    assemble_kernel<<<cdiv(N64, 256), 256, 0, stream>>>(x, h_l, mlp_w, mlp_b, tbuf);
    // 3. edge embedding e = edge_attr @ edge_w.T + edge_b
    linear_kernel<<<cdiv(NEDGE * 32, 256), 256, 0, stream>>>(edge_attr, edge_w, edge_b, eemb, NEDGE, 4, 32, 0);
    // 4. pre-permute per-layer w2/b2 into WMMA fragment order
    build_w2b_kernel<<<cdiv(4 * 64 * 4 * 32 * 16 + 4 * 2 * 4 * 32 * 16, 256), 256, 0, stream>>>(nn2_w, nn2_b, W2B, B2rB);

    // 5. four NNConv layers
    for (int l = 0; l < 4; ++l) {
        if (l > 0)
            ln_relu_kernel<<<NNODE, 32, 0, stream>>>(hbuf, ln_g + l * 64, ln_b + l * 64, tbuf, 1);
        linear_kernel<<<cdiv(NEDGE * 32, 256), 256, 0, stream>>>(eemb, nn1_w + l * 1024, nn1_b + l * 32, ew, NEDGE, 32, 32, 0);
        hipMemsetAsync(agg, 0, (size_t)N64 * 4, stream);
        nnconv_msg_kernel<<<NEDGE / 16, 32, 0, stream>>>(
            tbuf, ew, src, dst,
            W2B + (size_t)l * 64 * 4 * 32 * 16,
            B2rB + (size_t)l * 2 * 4 * 32 * 16, agg);
        combine_kernel<<<cdiv(N64, 256), 256, 0, stream>>>(tbuf, agg, root_w + l * 4096, gcn_b + l * 64, hbuf, l > 0);
    }
    // 6. final LN+ReLU (layer-0 params), gather nucleotide rows
    ln_relu_kernel<<<NNODE, 32, 0, stream>>>(hbuf, ln_g, ln_b, tbuf, 1);
    gather_nuc_kernel<<<cdiv(NS * 64, 256), 256, 0, stream>>>(tbuf, h_t);

    // 7. transformer block
    linear_kernel<<<cdiv(NS * 192, 256), 256, 0, stream>>>(h_t, inpj_w, inpj_b, qkv, NS, 64, 192, 0);
    attn_kernel<<<cdiv(SEQ * 4 * 16, 256), 256, 0, stream>>>(qkv, obuf);
    linear_kernel<<<cdiv(NS * 64, 256), 256, 0, stream>>>(obuf, outp_w, outp_b, tmp1, NS, 64, 64, 0);
    add_kernel<<<cdiv(NS * 64, 256), 256, 0, stream>>>(h_t, tmp1, tmp2, NS * 64);
    ln_relu_kernel<<<NS, 32, 0, stream>>>(tmp2, ln1_g, ln1_b, obuf, 0);   // obuf = h after LN1
    linear_kernel<<<cdiv(NS * 64, 256), 256, 0, stream>>>(obuf, ff1_w, ff1_b, tmp1, NS, 64, 64, 1);
    linear_kernel<<<cdiv(NS * 64, 256), 256, 0, stream>>>(tmp1, ff2_w, ff2_b, tmp2, NS, 64, 64, 0);
    add_kernel<<<cdiv(NS * 64, 256), 256, 0, stream>>>(obuf, tmp2, tmp1, NS * 64);
    ln_relu_kernel<<<NS, 32, 0, stream>>>(tmp1, ln2_g, ln2_b, hs2, 0);

    // 8. BiLSTM #2 (rows b*107+s, width 64)
    lstm_kernel<<<BATCH * 2, 128, 0, stream>>>(hs2, SEQ, 64, l2_wih, l2_whh, l2_bih, l2_bhh, h2);
    // 9. final projection to 3 outputs
    linear_kernel<<<cdiv(NS * 3, 256), 256, 0, stream>>>(h2, lin_w, lin_b, outp, NS, 64, 3, 0);
}
